// EfficientREN_50483045598036
// MI455X (gfx1250) — matrix operand order repaired and checked by reading, not compiled, not run on hardware
//
#include <hip/hip_runtime.h>
#include <hip/hip_bf16.h>
#include <math.h>

#define N_NODES 50000
#define E_EDGES 800000
#define ESM 1280
#define HID 256
#define HEADS 8
#define HD 32
#define L_LAYERS 3
#define BSZ 4096
#define MSITES 8
#define AA 21
#define DIMC (HID + AA) /* 277 */
#define DIMP 280        /* padded LDS row stride (16B multiple) */
#define NEG_SLOPE 0.2f
#define LN_EPS 1e-5f

#define USE_ASYNC_GATHER 1   /* global_load_async_to_lds_b128 in predict */

typedef __attribute__((ext_vector_type(16))) __bf16 v16bf;
typedef __attribute__((ext_vector_type(8)))  float  v8f;

// ---- monotone float<->uint mapping for atomicMax on floats of any sign ----
__device__ __forceinline__ unsigned f2ord(float f) {
  unsigned u = __float_as_uint(f);
  return (u & 0x80000000u) ? ~u : (u | 0x80000000u);
}
__device__ __forceinline__ float ord2f(unsigned u) {
  u = (u & 0x80000000u) ? (u ^ 0x80000000u) : ~u;
  return __uint_as_float(u);
}

union bfpk { unsigned u; __bf16 h[2]; };

// =====================================================================
// Tiled bf16 WMMA GEMM:  C[M x N] = A[M x K] @ W[K x N] (+ bias)
// Block = 256 threads = 8 waves. Tile: 128(M) x 128(N), BK = 32.
// LDS tiles stored in *fragment order*: each lane's 16 bf16 halves are
// contiguous + 32B aligned -> 2x ds_load_b128 per fragment.
// Staging is two-phase: (1) issue ALL global loads unconditionally
// (row index clamped, zeroed by select) so LOADcnt batches them;
// (2) v_cvt_pk_bf16_f32 + permuted ds_store_b32.
// Compute preloads A + all 8 B fragments, then 8 back-to-back WMMAs
// behind a single DS wait. Ping-pong buffer: one s_barrier per k-step.
// =====================================================================

// A fragment half layout (ISA 7.12.2, 16-bit A 16x32):
//   lane = m + 16*((k%16)>=8), vgpr v: k = 2*(v&3)+pos + 16*(v>>2) + 8*(lane>=16)
// B fragment half layout (32x16): lane = n + 16*((k%16)>=8), half i = k%8 + 8*(k/16).
__device__ __forceinline__ void stage_tiles(
    const float* __restrict__ A, const float* __restrict__ W,
    unsigned* __restrict__ Asu, unsigned* __restrict__ Bsu,
    int tile_m, int tile_n, int k0, int Mdim, int Kdim, int Ndim, int tid)
{
  // ---------- phase 1: issue all global loads (no waits in between) ----------
  float2 av[8];
#pragma unroll
  for (int it = 0; it < 8; ++it) {
    int idx = tid + it * 256;
    int r = idx >> 4, c2 = idx & 15;          // A: row r, k = 2*c2
    int gr = tile_m + r;
    int grc = gr < Mdim ? gr : (Mdim - 1);    // clamp -> unconditional load
    av[it] = *(const float2*)&A[(size_t)grc * Kdim + k0 + 2 * c2];
  }
  float bv0[8], bv1[8];
#pragma unroll
  for (int it = 0; it < 8; ++it) {
    int idx = tid + it * 256;
    int n = idx & 127, k = (idx >> 7) << 1;   // B: col n, k pair
    const float* p = &W[(size_t)(k0 + k) * Ndim + tile_n + n];
    bv0[it] = p[0]; bv1[it] = p[Ndim];
  }
  // ---------- phase 2: convert + fragment-order LDS stores ----------
#pragma unroll
  for (int it = 0; it < 8; ++it) {
    int idx = tid + it * 256;
    int r = idx >> 4, c2 = idx & 15;
    bool ok = (tile_m + r) < Mdim;
    bfpk pk;
    pk.h[0] = (__bf16)(ok ? av[it].x : 0.0f);
    pk.h[1] = (__bf16)(ok ? av[it].y : 0.0f);
    int m = r & 15, wt = r >> 4;
    int lane = m + (((c2 >> 2) & 1) << 4);    // (k%16)>=8 ?
    int vg = (c2 & 3) + ((c2 >> 3) << 2);     // (k%8)/2 + 4*(k/16)
    Asu[(wt * 32 + lane) * 8 + vg] = pk.u;
  }
#pragma unroll
  for (int it = 0; it < 8; ++it) {
    int idx = tid + it * 256;
    int n = idx & 127, k = (idx >> 7) << 1;
    bfpk pk; pk.h[0] = (__bf16)bv0[it]; pk.h[1] = (__bf16)bv1[it];
    int j = n >> 4;
    int lane = (n & 15) + (((k >> 3) & 1) << 4);
    Bsu[(j * 32 + lane) * 8 + ((k & 7) >> 1) + ((k >> 4) << 2)] = pk.u;
  }
}

__global__ __launch_bounds__(256) void gemm_bias_wmma(
    const float* __restrict__ A, const float* __restrict__ W,
    const float* __restrict__ bias, float* __restrict__ C,
    int Mdim, int Kdim, int Ndim)
{
  // fragment-ordered LDS: A: 8 waves * 32 lanes * 8 dwords; B: 8 frags * 32 * 8
  __shared__ __attribute__((aligned(32))) unsigned Asu[2][8 * 32 * 8];  // 2 x 8 KB
  __shared__ __attribute__((aligned(32))) unsigned Bsu[2][8 * 32 * 8];  // 2 x 8 KB

  const int tid    = threadIdx.x;
  const int lane   = tid & 31;
  const int wv     = tid >> 5;
  const int tile_m = blockIdx.y * 128;
  const int tile_n = blockIdx.x * 128;

  v8f acc[8] = {};

  stage_tiles(A, W, Asu[0], Bsu[0], tile_m, tile_n, 0, Mdim, Kdim, Ndim, tid);
  __syncthreads();

  int cur = 0;
  for (int k0 = 0; k0 < Kdim; k0 += 32) {
    // issue next slab's global loads + fragment-order LDS stores first,
    // so VMEM latency overlaps the WMMA sequence below
    if (k0 + 32 < Kdim)
      stage_tiles(A, W, Asu[cur ^ 1], Bsu[cur ^ 1],
                  tile_m, tile_n, k0 + 32, Mdim, Kdim, Ndim, tid);
    // keep L2 -> WGP pipe primed two slabs ahead (global_prefetch_b8)
    if (k0 + 64 < Kdim) {
      int gr = tile_m + (tid & 127);
      if (gr < Mdim) __builtin_prefetch(&A[(size_t)gr * Kdim + k0 + 64], 0, 1);
      if (tid < 32)  __builtin_prefetch(&W[(size_t)(k0 + 64 + tid) * Ndim + tile_n], 0, 1);
    }

    // preload A + all 8 B fragments (18x ds_load_b128), then run the 8
    // WMMAs back-to-back behind a single DS wait
    v16bf a = *(const v16bf*)(Asu[cur] + (wv * 32 + lane) * 8);
    v16bf bf[8];
#pragma unroll
    for (int j = 0; j < 8; ++j)
      bf[j] = *(const v16bf*)(Bsu[cur] + (j * 32 + lane) * 8);
#pragma unroll
    for (int j = 0; j < 8; ++j)
      acc[j] = __builtin_amdgcn_wmma_f32_16x16x32_bf16(
          false, a, false, bf[j], (short)0, acc[j], false, false);
    __syncthreads();
    cur ^= 1;
  }

  // --- store: C/D layout: lane l -> N = l%16, VGPR r -> M = r + 8*(l/16) ---
#pragma unroll
  for (int j = 0; j < 8; ++j) {
#pragma unroll
    for (int r = 0; r < 8; ++r) {
      int row = tile_m + 16 * wv + r + ((lane >> 4) << 3);
      int col = tile_n + (j << 4) + (lane & 15);
      if (row < Mdim) {
        float v = acc[j][r];
        if (bias) v += bias[col];
        C[(size_t)row * Ndim + col] = v;
      }
    }
  }
}

// =====================================================================
// Per-layer scratch init: agg = 0, max = -inf (ordered-uint), ssum = 0
// =====================================================================
__global__ void init_layer_kernel(float* agg, unsigned* mbuf, float* ssum) {
  int i = blockIdx.x * blockDim.x + threadIdx.x;
  if (i < N_NODES * HID) agg[i] = 0.0f;
  if (i < N_NODES * HEADS) { mbuf[i] = 0x007FFFFFu; /* f2ord(-inf) */ ssum[i] = 0.0f; }
}

// =====================================================================
// Attention logits: al_s[n,h] = <xs[n,h,:], att_src[h,:]>, same for dst
// =====================================================================
__global__ void att_kernel(const float* __restrict__ xs,
                           const float* __restrict__ a_src,
                           const float* __restrict__ a_dst,
                           float* __restrict__ al_s, float* __restrict__ al_d) {
  int i = blockIdx.x * blockDim.x + threadIdx.x;
  if (i >= N_NODES * HEADS) return;
  int n = i / HEADS, hh = i % HEADS;
  const float* xr = xs + (size_t)n * HID + hh * HD;
  const float* as = a_src + hh * HD;
  const float* ad = a_dst + hh * HD;
  float s = 0.0f, d = 0.0f;
#pragma unroll
  for (int k = 0; k < HD; ++k) { s += xr[k] * as[k]; d += xr[k] * ad[k]; }
  al_s[i] = s; al_d[i] = d;
}

__device__ __forceinline__ void edge_nodes(const int* ei, int eidx, int& s, int& d) {
  if (eidx < E_EDGES) { s = ei[eidx]; d = ei[E_EDGES + eidx]; }
  else { s = d = eidx - E_EDGES; }   // self loops appended after edges
}
__device__ __forceinline__ float edge_logit(const float* al_s, const float* al_d,
                                            int s, int d, int hh) {
  float e = al_s[s * HEADS + hh] + al_d[d * HEADS + hh];
  return e > 0.0f ? e : NEG_SLOPE * e;  // leaky_relu(0.2)
}

// pass 1: segment max over dst (atomicMax on ordered uints)
__global__ void edge_max_kernel(const int* __restrict__ ei,
                                const float* __restrict__ al_s,
                                const float* __restrict__ al_d,
                                unsigned* __restrict__ mbuf) {
  int i = blockIdx.x * blockDim.x + threadIdx.x;
  if (i >= (E_EDGES + N_NODES) * HEADS) return;
  int eidx = i / HEADS, hh = i % HEADS;
  int s, d; edge_nodes(ei, eidx, s, d);
  atomicMax(&mbuf[d * HEADS + hh], f2ord(edge_logit(al_s, al_d, s, d, hh)));
}

// pass 2: segment sum of exp(e - m)
__global__ void edge_sum_kernel(const int* __restrict__ ei,
                                const float* __restrict__ al_s,
                                const float* __restrict__ al_d,
                                const unsigned* __restrict__ mbuf,
                                float* __restrict__ ssum) {
  int i = blockIdx.x * blockDim.x + threadIdx.x;
  if (i >= (E_EDGES + N_NODES) * HEADS) return;
  int eidx = i / HEADS, hh = i % HEADS;
  int s, d; edge_nodes(ei, eidx, s, d);
  float e = edge_logit(al_s, al_d, s, d, hh);
  atomicAdd(&ssum[d * HEADS + hh], expf(e - ord2f(mbuf[d * HEADS + hh])));
}

// pass 3: agg[dst,h,:] += w * xs[src,h,:]  (atomics land in L2; agg fits in 192MB L2)
__global__ void edge_msg_kernel(const int* __restrict__ ei,
                                const float* __restrict__ al_s,
                                const float* __restrict__ al_d,
                                const unsigned* __restrict__ mbuf,
                                const float* __restrict__ ssum,
                                const float* __restrict__ xs,
                                float* __restrict__ agg) {
  int i = blockIdx.x * blockDim.x + threadIdx.x;
  if (i >= (E_EDGES + N_NODES) * HEADS) return;
  int eidx = i / HEADS, hh = i % HEADS;
  int s, d; edge_nodes(ei, eidx, s, d);
  float e = edge_logit(al_s, al_d, s, d, hh);
  float w = expf(e - ord2f(mbuf[d * HEADS + hh])) / ssum[d * HEADS + hh];
  const float* xr = xs + (size_t)s * HID + hh * HD;
  float* ar = agg + (size_t)d * HID + hh * HD;
#pragma unroll
  for (int k = 0; k < HD; ++k) atomicAdd(&ar[k], w * xr[k]);
}

// =====================================================================
// h = LayerNorm(elu(agg + gat_b) + h) * g + b ; one wave32 per node
// =====================================================================
__global__ __launch_bounds__(256) void post_kernel(const float* __restrict__ agg,
                                                   const float* __restrict__ gbias,
                                                   const float* __restrict__ g,
                                                   const float* __restrict__ bb,
                                                   float* __restrict__ h) {
  int lane = threadIdx.x & 31, wv = threadIdx.x >> 5;
  int node = blockIdx.x * 8 + wv;
  if (node >= N_NODES) return;
  float v[8]; float sum = 0.0f;
#pragma unroll
  for (int i = 0; i < 8; ++i) {
    int d = lane + 32 * i;
    float t = agg[(size_t)node * HID + d] + gbias[d];
    t = t > 0.0f ? t : (expf(t) - 1.0f);            // elu
    t += h[(size_t)node * HID + d];                 // residual
    v[i] = t; sum += t;
  }
#pragma unroll
  for (int o = 16; o; o >>= 1) sum += __shfl_xor(sum, o, 32);
  float mu = sum * (1.0f / HID);
  float vs = 0.0f;
#pragma unroll
  for (int i = 0; i < 8; ++i) { float dd = v[i] - mu; vs += dd * dd; }
#pragma unroll
  for (int o = 16; o; o >>= 1) vs += __shfl_xor(vs, o, 32);
  float rs = rsqrtf(vs * (1.0f / HID) + LN_EPS);
#pragma unroll
  for (int i = 0; i < 8; ++i) {
    int d = lane + 32 * i;
    h[(size_t)node * HID + d] = (v[i] - mu) * rs * g[d] + bb[d];
  }
}

// =====================================================================
// predict_batch: one block (128 threads) per batch element.
// Site-embedding gather is a conversion-free strided copy global->LDS:
// use GLOBAL_LOAD_ASYNC_TO_LDS_B128 (ASYNCcnt path, no VGPR round trip).
// =====================================================================
__global__ __launch_bounds__(128) void predict_kernel(
    const float* __restrict__ h, const int* __restrict__ sites,
    const float* __restrict__ mut, const unsigned char* __restrict__ mask,
    const float* __restrict__ pq_w, const float* __restrict__ pq_b,
    const float* __restrict__ w1, const float* __restrict__ b1,
    const float* __restrict__ w2, const float* __restrict__ b2,
    float* __restrict__ out) {
  __shared__ __attribute__((aligned(16))) float comb[MSITES][DIMP];
  __shared__ float sc[MSITES];
  __shared__ float wgt[MSITES];
  __shared__ float pooled[DIMC];
  __shared__ float red[128];
  int b = blockIdx.x, t = threadIdx.x;

#if USE_ASYNC_GATHER
  // embeddings: 8 sites x 256 floats = 512 x 16B async chunks, 4 per thread
#pragma unroll
  for (int it = 0; it < 4; ++it) {
    int i = t + it * 128;
    int m = i >> 6, j4 = i & 63;   // j4: 16B chunk within the 1KB row
    unsigned long long ga = (unsigned long long)(uintptr_t)
        &h[(size_t)sites[b * MSITES + m] * HID + j4 * 4];
    unsigned ldsoff = (unsigned)(uintptr_t)&comb[m][j4 * 4];
    asm volatile("global_load_async_to_lds_b128 %0, %1, off"
                 :: "v"(ldsoff), "v"(ga) : "memory");
  }
  for (int i = t; i < MSITES * AA; i += 128) {
    int m = i / AA, j = i % AA;
    comb[m][HID + j] = mut[(size_t)(b * MSITES + m) * AA + j];
  }
  asm volatile("s_wait_asynccnt 0x0" ::: "memory");
  __syncthreads();
#else
  for (int i = t; i < MSITES * DIMC; i += 128) {
    int m = i / DIMC, j = i % DIMC;
    comb[m][j] = (j < HID) ? h[(size_t)sites[b * MSITES + m] * HID + j]
                           : mut[(size_t)(b * MSITES + m) * AA + (j - HID)];
  }
  __syncthreads();
#endif

  if (t < MSITES) {
    float s = pq_b[0];
    for (int j = 0; j < DIMC; ++j) s += comb[t][j] * pq_w[j];
    sc[t] = mask[b * MSITES + t] ? s : -__builtin_inff();
  }
  __syncthreads();

  if (t == 0) {
    float mx = -__builtin_inff();
    for (int m = 0; m < MSITES; ++m) mx = fmaxf(mx, sc[m]);
    float ss = 0.0f;
    for (int m = 0; m < MSITES; ++m) { wgt[m] = expf(sc[m] - mx); ss += wgt[m]; }
    for (int m = 0; m < MSITES; ++m) wgt[m] /= ss;
  }
  __syncthreads();

  for (int j = t; j < DIMC; j += 128) {
    float p = 0.0f;
    for (int m = 0; m < MSITES; ++m) p += wgt[m] * comb[m][j];
    pooled[j] = p;
  }
  __syncthreads();

  float hv = b1[t];
  for (int j = 0; j < DIMC; ++j) hv += pooled[j] * w1[j * (HID / 2) + t];
  hv = hv > 0.0f ? hv : 0.0f;            // relu
  red[t] = hv * w2[t];
  __syncthreads();
  for (int o = 64; o; o >>= 1) { if (t < o) red[t] += red[t + o]; __syncthreads(); }
  if (t == 0) out[b] = red[0] + b2[0];
}

// =====================================================================
extern "C" void kernel_launch(void* const* d_in, const int* in_sizes, int n_in,
                              void* d_out, int out_size, void* d_ws, size_t ws_size,
                              hipStream_t stream) {
  (void)in_sizes; (void)n_in; (void)out_size; (void)ws_size;

  const float* x       = (const float*)d_in[0];
  const int*   ei      = (const int*)d_in[1];
  const int*   sites   = (const int*)d_in[2];
  const float* mut     = (const float*)d_in[3];
  const unsigned char* mask = (const unsigned char*)d_in[4];
  const float* proj_w  = (const float*)d_in[5];
  const float* proj_b  = (const float*)d_in[6];
  const float* lin_w   = (const float*)d_in[7];
  const float* att_src = (const float*)d_in[8];
  const float* att_dst = (const float*)d_in[9];
  const float* gat_b   = (const float*)d_in[10];
  const float* ln_g    = (const float*)d_in[11];
  const float* ln_b    = (const float*)d_in[12];
  const float* pq_w    = (const float*)d_in[13];
  const float* pq_b    = (const float*)d_in[14];
  const float* vh_w1   = (const float*)d_in[15];
  const float* vh_b1   = (const float*)d_in[16];
  const float* vh_w2   = (const float*)d_in[17];
  const float* vh_b2   = (const float*)d_in[18];
  float* out = (float*)d_out;

  // workspace carve-up (~160 MB of floats)
  float* ws = (float*)d_ws;
  float* h    = ws;   ws += (size_t)N_NODES * HID;
  float* xsb  = ws;   ws += (size_t)N_NODES * HID;
  float* agg  = ws;   ws += (size_t)N_NODES * HID;
  float* al_s = ws;   ws += (size_t)N_NODES * HEADS;
  float* al_d = ws;   ws += (size_t)N_NODES * HEADS;
  unsigned* mbuf = (unsigned*)ws; ws += (size_t)N_NODES * HEADS;
  float* ssum = ws;

  const dim3 blk(256);
  const dim3 gemm_grid(HID / 128, (N_NODES + 127) / 128);
  const int ecnt = (E_EDGES + N_NODES) * HEADS;

  // h = x @ proj_w + proj_b
  gemm_bias_wmma<<<gemm_grid, blk, 0, stream>>>(x, proj_w, proj_b, h,
                                                N_NODES, ESM, HID);

  for (int l = 0; l < L_LAYERS; ++l) {
    init_layer_kernel<<<(N_NODES * HID + 255) / 256, blk, 0, stream>>>(agg, mbuf, ssum);
    gemm_bias_wmma<<<gemm_grid, blk, 0, stream>>>(h, lin_w + (size_t)l * HID * HID,
                                                  nullptr, xsb, N_NODES, HID, HID);
    att_kernel<<<(N_NODES * HEADS + 255) / 256, blk, 0, stream>>>(
        xsb, att_src + l * HEADS * HD, att_dst + l * HEADS * HD, al_s, al_d);
    edge_max_kernel<<<(ecnt + 255) / 256, blk, 0, stream>>>(ei, al_s, al_d, mbuf);
    edge_sum_kernel<<<(ecnt + 255) / 256, blk, 0, stream>>>(ei, al_s, al_d, mbuf, ssum);
    edge_msg_kernel<<<(ecnt + 255) / 256, blk, 0, stream>>>(ei, al_s, al_d, mbuf, ssum, xsb, agg);
    post_kernel<<<(N_NODES + 7) / 8, blk, 0, stream>>>(
        agg, gat_b + l * HID, ln_g + l * HID, ln_b + l * HID, h);
  }

  predict_kernel<<<BSZ, dim3(128), 0, stream>>>(h, sites, mut, mask, pq_w, pq_b,
                                                vh_w1, vh_b1, vh_w2, vh_b2, out);
}
